// MambaBlock_75600014344464
// MI455X (gfx1250) — compile-verified, matrix operand order checked by reading, not run on hardware
//
#include <hip/hip_runtime.h>

#define HIDDEN  1024
#define D_INNER 2048
#define DT_RANK 64
#define STATE   16
#define KCONV   4
#define BATCH   2
#define SEQ     1024
#define MTOT    (BATCH * SEQ)   // 2048 rows in all GEMMs

#define CH      16              // scan chunks per sequence
#define CL      (SEQ / CH)      // 64 timesteps per chunk

typedef __attribute__((ext_vector_type(16))) __bf16 bf16x16;
typedef __attribute__((ext_vector_type(8)))  __bf16 bf16x8;
typedef __attribute__((ext_vector_type(4)))  __bf16 bf16x4;
typedef __attribute__((ext_vector_type(8)))  float  floatx8;

// ---------------------------------------------------------------------------
// fp32 -> bf16 pre-conversion (4 elements/thread, n divisible by 4)
// ---------------------------------------------------------------------------
__global__ void k_cvt_bf16(const float* __restrict__ in, __bf16* __restrict__ out,
                           int n4) {
  int i = blockIdx.x * blockDim.x + threadIdx.x;
  if (i >= n4) return;
  float4 f = ((const float4*)in)[i];
  bf16x4 o;
  o[0] = (__bf16)f.x; o[1] = (__bf16)f.y; o[2] = (__bf16)f.z; o[3] = (__bf16)f.w;
  ((bf16x4*)out)[i] = o;
}

// ---------------------------------------------------------------------------
// bf16 A/B fragment loader for V_WMMA_F32_16X16X32_BF16 (ISA 7.12.2):
// lane l: row = l&15, half h = l>>4. 16 bf16 per lane =
//   K in [k + h*8, k + h*8 + 8)  U  [k + 16 + h*8, k + 16 + h*8 + 8)
// Two 16-byte contiguous loads, no conversion.
// ---------------------------------------------------------------------------
__device__ __forceinline__ bf16x16 load_frag(const __bf16* __restrict__ row,
                                             int kbase, int half) {
  const __bf16* p = row + kbase + (half << 3);
  bf16x8 lo = *(const bf16x8*)p;
  bf16x8 hi = *(const bf16x8*)(p + 16);
  bf16x16 v;
#pragma unroll
  for (int i = 0; i < 8; ++i) { v[i] = lo[i]; v[i + 8] = hi[i]; }
  return v;
}

// ---------------------------------------------------------------------------
// 1) in_proj: xr = x @ in_proj_w^T, split into xs_pre / res.
//    M=2048, N=4096, K=1024.  One wave per 16x64 output strip.
// ---------------------------------------------------------------------------
__global__ void __launch_bounds__(256)
k_in_proj(const __bf16* __restrict__ X, const __bf16* __restrict__ W,
          float* __restrict__ xs_pre, float* __restrict__ res) {
  const int lane = threadIdx.x & 31;
  const int job  = (int)((blockIdx.x * blockDim.x + threadIdx.x) >> 5);
  const int NJ   = (2 * D_INNER) / 64;          // 64 strips along N
  const int m0   = (job / NJ) * 16;
  const int n0   = (job % NJ) * 64;
  const int r    = lane & 15, half = lane >> 4;

  const __bf16* arow = X + (size_t)(m0 + r) * HIDDEN;
  const __bf16* brow = W + (size_t)(n0 + r) * HIDDEN;

  floatx8 acc0 = {}, acc1 = {}, acc2 = {}, acc3 = {};
  for (int k = 0; k < HIDDEN; k += 32) {
    bf16x16 a  = load_frag(arow, k, half);
    bf16x16 b0 = load_frag(brow,                       k, half);
    bf16x16 b1 = load_frag(brow + (size_t)16 * HIDDEN, k, half);
    bf16x16 b2 = load_frag(brow + (size_t)32 * HIDDEN, k, half);
    bf16x16 b3 = load_frag(brow + (size_t)48 * HIDDEN, k, half);
    acc0 = __builtin_amdgcn_wmma_f32_16x16x32_bf16(false, a, false, b0, (short)0, acc0, false, false);
    acc1 = __builtin_amdgcn_wmma_f32_16x16x32_bf16(false, a, false, b1, (short)0, acc1, false, false);
    acc2 = __builtin_amdgcn_wmma_f32_16x16x32_bf16(false, a, false, b2, (short)0, acc2, false, false);
    acc3 = __builtin_amdgcn_wmma_f32_16x16x32_bf16(false, a, false, b3, (short)0, acc3, false, false);
  }

  const int nn    = lane & 15;
  const int mbase = m0 + half * 8;
  float* outp = (n0 < D_INNER) ? xs_pre : res;   // strip never straddles the split
  const int nc = (n0 < D_INNER) ? n0 : (n0 - D_INNER);
#pragma unroll
  for (int rr = 0; rr < 8; ++rr) {
    size_t ro = (size_t)(mbase + rr) * D_INNER;
    outp[ro + nc +  0 + nn] = acc0[rr];
    outp[ro + nc + 16 + nn] = acc1[rr];
    outp[ro + nc + 32 + nn] = acc2[rr];
    outp[ro + nc + 48 + nn] = acc3[rr];
  }
}

// ---------------------------------------------------------------------------
// 2) causal depthwise conv1d (pad left 3) + bias + SiLU -> xs (f32) + xsb (bf16)
// ---------------------------------------------------------------------------
__global__ void k_conv_silu(const float* __restrict__ xs_pre,
                            const float* __restrict__ cw, const float* __restrict__ cb,
                            float* __restrict__ xs, __bf16* __restrict__ xsb) {
  int idx = blockIdx.x * blockDim.x + threadIdx.x;
  if (idx >= MTOT * D_INNER) return;
  int d = idx % D_INNER;
  int m = idx / D_INNER;
  int l = m % SEQ;
  int b = m / SEQ;
  float acc = cb[d];
#pragma unroll
  for (int j = 0; j < KCONV; ++j) {
    int t = l - (KCONV - 1) + j;
    if (t >= 0)
      acc += cw[d * KCONV + j] * xs_pre[((size_t)(b * SEQ + t)) * D_INNER + d];
  }
  float s = acc / (1.0f + expf(-acc));          // SiLU
  xs[idx]  = s;
  xsb[idx] = (__bf16)s;
}

// ---------------------------------------------------------------------------
// 3) x_proj: x_dbl = xs @ x_proj_w^T  -> dt(64, bf16) | B(16, f32) | C(16, f32)
//    M=2048, N=96, K=2048.  One wave per 16x96 row strip (6 tiles).
// ---------------------------------------------------------------------------
__global__ void __launch_bounds__(256)
k_x_proj(const __bf16* __restrict__ XS, const __bf16* __restrict__ W,
         __bf16* __restrict__ dtm, float* __restrict__ Bm, float* __restrict__ Cm) {
  const int lane = threadIdx.x & 31;
  const int job  = (int)((blockIdx.x * blockDim.x + threadIdx.x) >> 5);  // 0..127
  const int m0   = job * 16;
  const int r    = lane & 15, half = lane >> 4;
  const __bf16* arow = XS + (size_t)(m0 + r) * D_INNER;
  const __bf16* brow = W  + (size_t)r * D_INNER;

  floatx8 acc[6];
#pragma unroll
  for (int i = 0; i < 6; ++i) { floatx8 z = {}; acc[i] = z; }

  for (int k = 0; k < D_INNER; k += 32) {
    bf16x16 a = load_frag(arow, k, half);
#pragma unroll
    for (int nt = 0; nt < 6; ++nt) {
      bf16x16 b = load_frag(brow + (size_t)nt * 16 * D_INNER, k, half);
      acc[nt] = __builtin_amdgcn_wmma_f32_16x16x32_bf16(false, a, false, b, (short)0, acc[nt], false, false);
    }
  }

  const int nn = lane & 15;
  const int mbase = m0 + half * 8;
#pragma unroll
  for (int rr = 0; rr < 8; ++rr) {
    int m = mbase + rr;
#pragma unroll
    for (int nt = 0; nt < 6; ++nt) {
      int n = nt * 16 + nn;
      float v = acc[nt][rr];
      if (n < DT_RANK)                dtm[(size_t)m * DT_RANK + n] = (__bf16)v;
      else if (n < DT_RANK + STATE)   Bm[(size_t)m * STATE + (n - DT_RANK)] = v;
      else                            Cm[(size_t)m * STATE + (n - DT_RANK - STATE)] = v;
    }
  }
}

// ---------------------------------------------------------------------------
// 4) dt_proj + softplus: delta = softplus(dt @ dt_proj_w^T + b)
//    M=2048, N=2048, K=64.
// ---------------------------------------------------------------------------
__global__ void __launch_bounds__(256)
k_dt_proj(const __bf16* __restrict__ DT, const __bf16* __restrict__ W,
          const float* __restrict__ bias, float* __restrict__ delta) {
  const int lane = threadIdx.x & 31;
  const int job  = (int)((blockIdx.x * blockDim.x + threadIdx.x) >> 5);
  const int NJ   = D_INNER / 64;                 // 32
  const int m0   = (job / NJ) * 16;
  const int n0   = (job % NJ) * 64;
  const int r    = lane & 15, half = lane >> 4;
  const __bf16* arow = DT + (size_t)(m0 + r) * DT_RANK;
  const __bf16* brow = W  + (size_t)(n0 + r) * DT_RANK;

  floatx8 acc[4];
#pragma unroll
  for (int i = 0; i < 4; ++i) { floatx8 z = {}; acc[i] = z; }

  for (int k = 0; k < DT_RANK; k += 32) {
    bf16x16 a = load_frag(arow, k, half);
#pragma unroll
    for (int nt = 0; nt < 4; ++nt) {
      bf16x16 b = load_frag(brow + (size_t)nt * 16 * DT_RANK, k, half);
      acc[nt] = __builtin_amdgcn_wmma_f32_16x16x32_bf16(false, a, false, b, (short)0, acc[nt], false, false);
    }
  }

  const int nn = lane & 15;
  const int mbase = m0 + half * 8;
#pragma unroll
  for (int rr = 0; rr < 8; ++rr) {
    size_t ro = (size_t)(mbase + rr) * D_INNER;
#pragma unroll
    for (int nt = 0; nt < 4; ++nt) {
      int n = n0 + nt * 16 + nn;
      float v = acc[nt][rr] + bias[n];
      // numerically-stable softplus: max(v,0) + log1p(exp(-|v|))
      delta[ro + n] = fmaxf(v, 0.0f) + log1pf(expf(-fabsf(v)));
    }
  }
}

// ---------------------------------------------------------------------------
// 5) chunked parallel selective scan.  h <- exp(delta*a)*h + delta*u*B is an
//    affine recurrence per (b,d,n); a chunk's transition is exp(a * sum(delta))
//    (product of exponentials == exp of sum), so the prefix pass is cheap.
//
//    scanA: (b,d,chunk) threads, h from 0, emit hpart[16] + sum(delta).
//    scanB: (b,d) threads, 16-chunk prefix; hpart overwritten with entry state.
//    scanC: (b,d,chunk) threads, re-scan from entry state, emit gated bf16 y.
//
//    B/C chunk rows (4KB each) staged into LDS via global_load_async_to_lds.
// ---------------------------------------------------------------------------
__global__ void __launch_bounds__(256)
k_scanA(const float* __restrict__ xs, const float* __restrict__ delta,
        const float* __restrict__ Bm, const float* __restrict__ A_log,
        float* __restrict__ hpart, float* __restrict__ sdelta) {
  __shared__ float sB[CL * STATE];               // 4KB
  const int tid  = threadIdx.x;
  const int dblk = blockIdx.x & 7;               // 8 d-blocks of 256
  const int c    = (blockIdx.x >> 3) & (CH - 1);
  const int b    = blockIdx.x >> 7;
  const int d    = dblk * 256 + tid;
  const size_t m0 = (size_t)b * SEQ + (size_t)c * CL;

  // async-stage B chunk: 256 threads x 16B = 4KB
  {
    const float* g = Bm + m0 * STATE + tid * 4;
    unsigned l = (unsigned)(size_t)&sB[0] + tid * 16;   // low 32b == LDS offset
    asm volatile("global_load_async_to_lds_b128 %0, %1, off"
                 :: "v"(l), "v"((unsigned long long)(size_t)g) : "memory");
  }

  float a[STATE], h[STATE];
#pragma unroll
  for (int n = 0; n < STATE; ++n) {
    a[n] = -expf(A_log[d * STATE + n]);
    h[n] = 0.0f;
  }
  float sd = 0.0f;

  asm volatile("s_wait_asynccnt 0x0" ::: "memory");
  __syncthreads();

  for (int j = 0; j < CL; ++j) {
    size_t md = (m0 + j) * D_INNER + d;
    float dl = delta[md];
    sd += dl;
    float du = dl * xs[md];
#pragma unroll
    for (int n = 0; n < STATE; ++n)
      h[n] = expf(dl * a[n]) * h[n] + du * sB[j * STATE + n];
  }

  const size_t base = ((size_t)(b * CH + c) * STATE) * D_INNER + d;
#pragma unroll
  for (int n = 0; n < STATE; ++n) hpart[base + (size_t)n * D_INNER] = h[n];
  sdelta[(size_t)(b * CH + c) * D_INNER + d] = sd;
}

__global__ void __launch_bounds__(256)
k_scanB(const float* __restrict__ A_log, const float* __restrict__ sdelta,
        float* hpart) {                          // in: partials, out: entry states
  const int t = blockIdx.x * 256 + threadIdx.x;  // 4096 threads
  const int d = t & (D_INNER - 1);
  const int b = t >> 11;
  float a[STATE], h[STATE];
#pragma unroll
  for (int n = 0; n < STATE; ++n) {
    a[n] = -expf(A_log[d * STATE + n]);
    h[n] = 0.0f;
  }
  for (int c = 0; c < CH; ++c) {
    const size_t base = ((size_t)(b * CH + c) * STATE) * D_INNER + d;
    float sd = sdelta[(size_t)(b * CH + c) * D_INNER + d];
#pragma unroll
    for (int n = 0; n < STATE; ++n) {
      float E   = expf(a[n] * sd);               // chunk transition
      float tmp = hpart[base + (size_t)n * D_INNER];
      hpart[base + (size_t)n * D_INNER] = h[n];  // entry state for chunk c
      h[n] = E * h[n] + tmp;
    }
  }
}

__global__ void __launch_bounds__(256)
k_scanC(const float* __restrict__ xs, const float* __restrict__ delta,
        const float* __restrict__ Bm, const float* __restrict__ Cm,
        const float* __restrict__ A_log, const float* __restrict__ res,
        const float* __restrict__ hin, __bf16* __restrict__ ygb) {
  __shared__ float sB[CL * STATE];               // 4KB
  __shared__ float sC[CL * STATE];               // 4KB
  const int tid  = threadIdx.x;
  const int dblk = blockIdx.x & 7;
  const int c    = (blockIdx.x >> 3) & (CH - 1);
  const int b    = blockIdx.x >> 7;
  const int d    = dblk * 256 + tid;
  const size_t m0 = (size_t)b * SEQ + (size_t)c * CL;

  {
    const float* gB = Bm + m0 * STATE + tid * 4;
    const float* gC = Cm + m0 * STATE + tid * 4;
    unsigned lB = (unsigned)(size_t)&sB[0] + tid * 16;
    unsigned lC = (unsigned)(size_t)&sC[0] + tid * 16;
    asm volatile("global_load_async_to_lds_b128 %0, %1, off"
                 :: "v"(lB), "v"((unsigned long long)(size_t)gB) : "memory");
    asm volatile("global_load_async_to_lds_b128 %0, %1, off"
                 :: "v"(lC), "v"((unsigned long long)(size_t)gC) : "memory");
  }

  float a[STATE], h[STATE];
  const size_t base = ((size_t)(b * CH + c) * STATE) * D_INNER + d;
#pragma unroll
  for (int n = 0; n < STATE; ++n) {
    a[n] = -expf(A_log[d * STATE + n]);
    h[n] = hin[base + (size_t)n * D_INNER];      // true entry state
  }

  asm volatile("s_wait_asynccnt 0x0" ::: "memory");
  __syncthreads();

  for (int j = 0; j < CL; ++j) {
    size_t md = (m0 + j) * D_INNER + d;
    float dl = delta[md];
    float du = dl * xs[md];
    float y  = 0.0f;
#pragma unroll
    for (int n = 0; n < STATE; ++n) {
      h[n] = expf(dl * a[n]) * h[n] + du * sB[j * STATE + n];
      y   += h[n] * sC[j * STATE + n];
    }
    float rr = res[md];
    ygb[md] = (__bf16)(y * (rr / (1.0f + expf(-rr))));   // gate
  }
}

// ---------------------------------------------------------------------------
// 6) out_proj: out = y_gated @ out_proj_w^T.  M=2048, N=1024, K=2048.
// ---------------------------------------------------------------------------
__global__ void __launch_bounds__(256)
k_out_proj(const __bf16* __restrict__ Y, const __bf16* __restrict__ W,
           float* __restrict__ out) {
  const int lane = threadIdx.x & 31;
  const int job  = (int)((blockIdx.x * blockDim.x + threadIdx.x) >> 5);
  const int NJ   = HIDDEN / 64;                  // 16
  const int m0   = (job / NJ) * 16;
  const int n0   = (job % NJ) * 64;
  const int r    = lane & 15, half = lane >> 4;
  const __bf16* arow = Y + (size_t)(m0 + r) * D_INNER;
  const __bf16* brow = W + (size_t)(n0 + r) * D_INNER;

  floatx8 acc[4];
#pragma unroll
  for (int i = 0; i < 4; ++i) { floatx8 z = {}; acc[i] = z; }

  for (int k = 0; k < D_INNER; k += 32) {
    bf16x16 a = load_frag(arow, k, half);
#pragma unroll
    for (int nt = 0; nt < 4; ++nt) {
      bf16x16 b = load_frag(brow + (size_t)nt * 16 * D_INNER, k, half);
      acc[nt] = __builtin_amdgcn_wmma_f32_16x16x32_bf16(false, a, false, b, (short)0, acc[nt], false, false);
    }
  }

  const int nn = lane & 15;
  const int mbase = m0 + half * 8;
#pragma unroll
  for (int rr = 0; rr < 8; ++rr) {
    size_t ro = (size_t)(mbase + rr) * HIDDEN;
#pragma unroll
    for (int nt = 0; nt < 4; ++nt)
      out[ro + n0 + nt * 16 + nn] = acc[nt][rr];
  }
}

// ---------------------------------------------------------------------------
extern "C" void kernel_launch(void* const* d_in, const int* in_sizes, int n_in,
                              void* d_out, int out_size, void* d_ws, size_t ws_size,
                              hipStream_t stream) {
  (void)in_sizes; (void)n_in; (void)out_size; (void)ws_size;

  const float* x          = (const float*)d_in[0];
  const float* in_proj_w  = (const float*)d_in[1];
  const float* conv_w     = (const float*)d_in[2];
  const float* conv_b     = (const float*)d_in[3];
  const float* x_proj_w   = (const float*)d_in[4];
  const float* dt_proj_w  = (const float*)d_in[5];
  const float* dt_proj_b  = (const float*)d_in[6];
  const float* A_log      = (const float*)d_in[7];
  const float* out_proj_w = (const float*)d_in[8];
  float* out = (float*)d_out;

  // ---- workspace layout ----
  const size_t NBL = (size_t)MTOT * D_INNER;     // 4,194,304
  char* p = (char*)d_ws;
  auto take = [&](size_t bytes) { char* q = p; p += (bytes + 255) & ~(size_t)255; return q; };

  float* xs_pre = (float*)take(NBL * 4);                       // reused as delta
  float* res    = (float*)take(NBL * 4);
  float* xs     = (float*)take(NBL * 4);
  float* Bm     = (float*)take((size_t)MTOT * STATE * 4);
  float* Cm     = (float*)take((size_t)MTOT * STATE * 4);
  float* hpart  = (float*)take((size_t)BATCH * CH * STATE * D_INNER * 4);  // 4MB
  float* sdelta = (float*)take((size_t)BATCH * CH * D_INNER * 4);          // 256KB
  __bf16* xb     = (__bf16*)take((size_t)MTOT * HIDDEN * 2);
  __bf16* wb_in  = (__bf16*)take((size_t)2 * D_INNER * HIDDEN * 2);
  __bf16* wb_x   = (__bf16*)take((size_t)(DT_RANK + 2 * STATE) * D_INNER * 2);
  __bf16* wb_dt  = (__bf16*)take((size_t)D_INNER * DT_RANK * 2);
  __bf16* wb_out = (__bf16*)take((size_t)HIDDEN * D_INNER * 2);
  __bf16* xsb    = (__bf16*)take(NBL * 2);
  __bf16* dtm    = (__bf16*)take((size_t)MTOT * DT_RANK * 2);
  __bf16* ygb    = (__bf16*)take(NBL * 2);
  float* delta  = xs_pre;

  // 0) one-shot bf16 conversions of x and all GEMM weights
  auto cvt = [&](const float* src, __bf16* dst, size_t n) {
    int n4 = (int)(n / 4);
    k_cvt_bf16<<<(n4 + 255) / 256, 256, 0, stream>>>(src, dst, n4);
  };
  cvt(x,          xb,     (size_t)MTOT * HIDDEN);
  cvt(in_proj_w,  wb_in,  (size_t)2 * D_INNER * HIDDEN);
  cvt(x_proj_w,   wb_x,   (size_t)(DT_RANK + 2 * STATE) * D_INNER);
  cvt(dt_proj_w,  wb_dt,  (size_t)D_INNER * DT_RANK);
  cvt(out_proj_w, wb_out, (size_t)HIDDEN * D_INNER);

  // 1) in_proj: 8192 waves -> 1024 blocks x 8 waves
  k_in_proj<<<(MTOT / 16) * ((2 * D_INNER) / 64) / 8, 256, 0, stream>>>(
      xb, wb_in, xs_pre, res);

  // 2) conv + SiLU
  {
    int n = MTOT * D_INNER;
    k_conv_silu<<<(n + 255) / 256, 256, 0, stream>>>(xs_pre, conv_w, conv_b, xs, xsb);
  }

  // 3) x_proj: 128 waves -> 16 blocks
  k_x_proj<<<(MTOT / 16) / 8, 256, 0, stream>>>(xsb, wb_x, dtm, Bm, Cm);

  // 4) dt_proj + softplus: 4096 waves -> 512 blocks
  k_dt_proj<<<(MTOT / 16) * (D_INNER / 64) / 8, 256, 0, stream>>>(
      dtm, wb_dt, dt_proj_b, delta);

  // 5) chunked parallel scan: A (partials) -> B (prefix) -> C (emit, gated bf16)
  k_scanA<<<BATCH * CH * (D_INNER / 256), 256, 0, stream>>>(
      xs, delta, Bm, A_log, hpart, sdelta);
  k_scanB<<<(BATCH * D_INNER) / 256, 256, 0, stream>>>(A_log, sdelta, hpart);
  k_scanC<<<BATCH * CH * (D_INNER / 256), 256, 0, stream>>>(
      xs, delta, Bm, Cm, A_log, res, hpart, ygb);

  // 6) out_proj: 2048 waves -> 256 blocks
  k_out_proj<<<(MTOT / 16) * (HIDDEN / 64) / 8, 256, 0, stream>>>(ygb, wb_out, out);
}
